// MemoryModule_57552561766669
// MI455X (gfx1250) — matrix operand order repaired and checked by reading, not compiled, not run on hardware
//
#include <hip/hip_runtime.h>
#include <hip/hip_bf16.h>
#include <math.h>

// Problem sizes (fixed by the reference)
#define BQ   32768   // rows of X
#define MM   4096    // memory rows
#define DD   128     // feature dim
#define CHUNK 32     // memory rows per iteration (K=32 for wmma f16)

typedef _Float16 v16h __attribute__((ext_vector_type(16)));
typedef _Float16 v8h  __attribute__((ext_vector_type(8)));
typedef _Float16 v4h  __attribute__((ext_vector_type(4)));
typedef float    v8f  __attribute__((ext_vector_type(8)));
typedef float    v4f  __attribute__((ext_vector_type(4)));
typedef unsigned int u32x4 __attribute__((ext_vector_type(4)));
typedef unsigned int u32x8 __attribute__((ext_vector_type(8)));

static __device__ __forceinline__ v8f wmma_f16(v16h a, v16h b, v8f c) {
    return __builtin_amdgcn_wmma_f32_16x16x32_f16(
        false, a, false, b, (short)0, c, false, false);
}

// A-fragment (contiguous K order): lanes 0-15 hold K=0..15,
// lanes 16-31 hold K=16..31; two contiguous 16B groups per lane.
static __device__ __forceinline__ v16h load_afrag(const _Float16* row, int hi) {
    v8h a = *(const v8h*)(row + hi * 16);
    v8h b = *(const v8h*)(row + hi * 16 + 8);
    return __builtin_shufflevector(a, b, 0,1,2,3,4,5,6,7,8,9,10,11,12,13,14,15);
}

// GEMM2 A-fragment with permuted K enumeration (physical K -> m:
// [0,8):k, [8,16):k+16, [16,24):k-8, [24,32):k-8). Paired with the
// select-free P^T build below; bijective, so the contraction is unchanged.
// lo lanes: elems 0..7 <- m 0..7,  elems 8..15 <- m 24..31
// hi lanes: elems 0..7 <- m 8..15, elems 8..15 <- m 16..23
static __device__ __forceinline__ v16h load_afrag_perm(const _Float16* row, int hi) {
    v8h a = *(const v8h*)(row + (hi ? 8 : 0));
    v8h b = *(const v8h*)(row + (hi ? 16 : 24));
    return __builtin_shufflevector(a, b, 0,1,2,3,4,5,6,7,8,9,10,11,12,13,14,15);
}

static __device__ __forceinline__ float fast_exp2(float x) {
#if __has_builtin(__builtin_amdgcn_exp2f)
    return __builtin_amdgcn_exp2f(x);
#else
    return exp2f(x);
#endif
}

static __device__ __forceinline__ unsigned int pack2h(float a, float b) {
    union { _Float16 h[2]; unsigned int u; } t;
    t.h[0] = (_Float16)a; t.h[1] = (_Float16)b;
    return t.u;   // compiler fuses to v_cvt_pk_f16_f32
}

// ---------------------------------------------------------------------------
// TDM: 2-D tensor_load_to_lds (descriptor per ISA 08_async_tensor §8.3/8.4)
// ---------------------------------------------------------------------------
static __device__ __forceinline__ void tdm_load_2d(
        unsigned int lds_addr, const void* gaddr,
        unsigned int tile0, unsigned int tile1,
        unsigned int tdim0, unsigned int tdim1,
        unsigned long long stride0 /* data_size units */) {
    unsigned long long ga = (unsigned long long)(uintptr_t)gaddr;
    u32x4 g0;
    g0[0] = 1u;                                     // count=1, user mode
    g0[1] = lds_addr;                               // LDS byte address
    g0[2] = (unsigned int)ga;                       // global addr [31:0]
    g0[3] = (unsigned int)(ga >> 32) | 0x80000000u; // addr[56:32] | type=2
    u32x8 g1;
    g1[0] = (1u << 16);                             // data_size = 1 -> 2 bytes
    g1[1] = (tdim0 & 0xFFFFu) << 16;                // tensor_dim0[15:0]
    g1[2] = (tdim0 >> 16) | ((tdim1 & 0xFFFFu) << 16);
    g1[3] = (tdim1 >> 16) | (tile0 << 16);          // tile_dim0
    g1[4] = tile1;                                  // tile_dim1 (tile_dim2=0)
    g1[5] = (unsigned int)(stride0 & 0xFFFFFFFFu);
    g1[6] = (unsigned int)(stride0 >> 32);          // stride0[47:32], stride1=0
    g1[7] = 0u;
    asm volatile("tensor_load_to_lds %0, %1" :: "s"(g0), "s"(g1) : "memory");
}

static __device__ __forceinline__ void wait_tensorcnt0() {
#if __has_builtin(__builtin_amdgcn_s_wait_tensorcnt)
    __builtin_amdgcn_s_wait_tensorcnt(0);
#else
    asm volatile("s_wait_tensorcnt 0x0" ::: "memory");
#endif
}

// ---------------------------------------------------------------------------
// Prep: Mem f32 [M][D] -> ws: hi f16 [M][D], lo f16 [M][D], hi^T f16 [D][M]
// ---------------------------------------------------------------------------
__global__ __launch_bounds__(256)
void prep_kernel(const float* __restrict__ Mem,
                 _Float16* __restrict__ wsHi,
                 _Float16* __restrict__ wsLo,
                 _Float16* __restrict__ wsT) {
    int base = blockIdx.x * 256 + threadIdx.x;
    #pragma unroll
    for (int k = 0; k < 4; ++k) {
        int i = base + k * (MM * DD / 4);
        int m = i >> 7, d = i & 127;
        float v = Mem[i];
        _Float16 h = (_Float16)v;
        wsHi[i] = h;
        wsLo[i] = (_Float16)(v - (float)h);
        wsT[d * MM + m] = h;
    }
}

// ---------------------------------------------------------------------------
// Shared per-chunk compute core (GEMM1 + online softmax + GEMM2)
// sHiC/sLoC: [CHUNK][DD] f16 hi/lo ; sTC: [DD][CHUNK] f16 hi
// ---------------------------------------------------------------------------
static __device__ __forceinline__ void chunk_compute(
        const _Float16* sHiC, const _Float16* sLoC, const _Float16* sTC,
        const v16h* xh, const v16h* xl, int l16, int hi,
        float& run_m, float& run_s, v8f* o) {
    // ---- GEMM1: S^T = Mem_chunk x X^T (split-f16) ----
    v8f acc0, acc1;
    #pragma unroll
    for (int e = 0; e < 8; ++e) { acc0[e] = 0.0f; acc1[e] = 0.0f; }
    #pragma unroll
    for (int kc = 0; kc < 4; ++kc) {
        v16h a0h = load_afrag(sHiC + (l16     ) * DD + kc * 32, hi);
        v16h a0l = load_afrag(sLoC + (l16     ) * DD + kc * 32, hi);
        v16h a1h = load_afrag(sHiC + (l16 + 16) * DD + kc * 32, hi);
        v16h a1l = load_afrag(sLoC + (l16 + 16) * DD + kc * 32, hi);
        acc0 = wmma_f16(a0h, xh[kc], acc0);
        acc0 = wmma_f16(a0h, xl[kc], acc0);
        acc0 = wmma_f16(a0l, xh[kc], acc0);
        acc1 = wmma_f16(a1h, xh[kc], acc1);
        acc1 = wmma_f16(a1h, xl[kc], acc1);
        acc1 = wmma_f16(a1l, xh[kc], acc1);
    }

    float s[16];
    #pragma unroll
    for (int v = 0; v < 8; ++v) { s[v] = acc0[v]; s[8 + v] = acc1[v]; }

    // ---- online softmax (log2 domain), tree reductions ----
    float mx[8];
    #pragma unroll
    for (int j = 0; j < 8; ++j) mx[j] = fmaxf(s[2 * j], s[2 * j + 1]);
    #pragma unroll
    for (int w = 4; w >= 1; w >>= 1)
        #pragma unroll
        for (int j = 0; j < w; ++j) mx[j] = fmaxf(mx[j], mx[j + w]);
    float cm = fmaxf(mx[0], __shfl_xor(mx[0], 16, 32));
    float nm = fmaxf(run_m, cm);
    if (__any((int)(nm > run_m))) {     // rare after early chunks
        float scale = fast_exp2(run_m - nm);
        run_s *= scale;
        #pragma unroll
        for (int t = 0; t < 8; ++t)
            #pragma unroll
            for (int e = 0; e < 8; ++e) o[t][e] *= scale;
        run_m = nm;
    }
    float p[16];
    #pragma unroll
    for (int j = 0; j < 16; ++j) p[j] = fast_exp2(s[j] - run_m);
    float sm[8];
    #pragma unroll
    for (int j = 0; j < 8; ++j) sm[j] = p[2 * j] + p[2 * j + 1];
    #pragma unroll
    for (int w = 4; w >= 1; w >>= 1)
        #pragma unroll
        for (int j = 0; j < w; ++j) sm[j] += sm[j + w];
    run_s += sm[0] + __shfl_xor(sm[0], 16, 32);

    // ---- select-free P^T B-fragment (matches load_afrag_perm K order):
    // slots 0..7 = own p[0..7]; slots 8..15 = partner's p[8..15] elementwise.
    union { v16h h; unsigned int u[8]; } pf;
    #pragma unroll
    for (int j = 0; j < 4; ++j)
        pf.u[j] = pack2h(p[2 * j], p[2 * j + 1]);
    #pragma unroll
    for (int j = 4; j < 8; ++j) {
        unsigned int pk = pack2h(p[2 * j], p[2 * j + 1]);
        pf.u[j] = (unsigned int)__shfl_xor((int)pk, 16, 32);
    }

    // ---- GEMM2: O^T += Mem^T_chunk x P^T ----
    #pragma unroll
    for (int td = 0; td < 8; ++td) {
        v16h a = load_afrag_perm(sTC + (td * 16 + l16) * CHUNK, hi);
        o[td] = wmma_f16(a, pf.h, o[td]);
    }
}

static __device__ __forceinline__ void load_x_frags(
        const float* xrow, int hi, v16h* xh, v16h* xl) {
    const float LOG2E = 1.4426950408889634f;   // softmax done in exp2 domain
    #pragma unroll
    for (int kc = 0; kc < 4; ++kc) {
        const v4f* p4 = (const v4f*)(xrow + kc * 32 + hi * 16);
        #pragma unroll
        for (int g = 0; g < 4; ++g) {
            v4f f = p4[g];
            #pragma unroll
            for (int c = 0; c < 4; ++c) {
                int e = g * 4 + c;
                float v = f[c] * LOG2E;
                _Float16 h2 = (_Float16)v;
                xh[kc][e] = h2;
                xl[kc][e] = (_Float16)(v - (float)h2);
            }
        }
    }
}

static __device__ __forceinline__ void store_out(
        float* Out, int q, int hi, float run_s, const v8f* o) {
    float inv = 1.0f / run_s;
    float* orow = Out + (size_t)q * DD;
    #pragma unroll
    for (int td = 0; td < 8; ++td) {
        float* base = orow + td * 16 + hi * 8;
        v4f a, b;
        #pragma unroll
        for (int c = 0; c < 4; ++c) { a[c] = o[td][c] * inv; b[c] = o[td][4 + c] * inv; }
        *(v4f*)(base)     = a;
        *(v4f*)(base + 4) = b;
    }
}

// ---------------------------------------------------------------------------
// Fast path: TDM double-buffered flash attention read
// ---------------------------------------------------------------------------
__global__ __launch_bounds__(256, 1)
void attn_read_tdm(const float* __restrict__ X,
                   const _Float16* __restrict__ gHi,
                   const _Float16* __restrict__ gLo,
                   const _Float16* __restrict__ gT,
                   float* __restrict__ Out) {
    __shared__ __align__(16) _Float16 sHi[2][CHUNK * DD];
    __shared__ __align__(16) _Float16 sLo[2][CHUNK * DD];
    __shared__ __align__(16) _Float16 sT [2][DD * CHUNK];

    const int tid  = threadIdx.x;
    const int lane = tid & 31;
    const int wave = tid >> 5;
    const int l16  = lane & 15;
    const int hi   = lane >> 4;
    const int q    = blockIdx.x * 128 + wave * 16 + l16;

    v16h xh[4], xl[4];
    load_x_frags(X + (size_t)q * DD, hi, xh, xl);

    v8f o[8];
    #pragma unroll
    for (int t = 0; t < 8; ++t)
        #pragma unroll
        for (int e = 0; e < 8; ++e) o[t][e] = 0.0f;

    float run_m = -INFINITY, run_s = 0.0f;

    // Prime the pipeline: chunk 0 -> buffer 0 (wave 0 drives the TDM)
    if (wave == 0) {
        tdm_load_2d((unsigned int)(uintptr_t)&sHi[0][0], gHi, DD, CHUNK, DD, MM, DD);
        tdm_load_2d((unsigned int)(uintptr_t)&sLo[0][0], gLo, DD, CHUNK, DD, MM, DD);
        tdm_load_2d((unsigned int)(uintptr_t)&sT[0][0],  gT,  CHUNK, DD, MM, DD, MM);
    }

    const int NCH = MM / CHUNK;
    for (int i = 0; i < NCH; ++i) {
        const int cur = i & 1;
        if (wave == 0) wait_tensorcnt0();
        __syncthreads();   // buf[cur] ready; everyone done with buf[1-cur]

        if (i + 1 < NCH && wave == 0) {
            int cb = (i + 1) * CHUNK;
            tdm_load_2d((unsigned int)(uintptr_t)&sHi[1 - cur][0],
                        gHi + (size_t)cb * DD, DD, CHUNK, DD, MM, DD);
            tdm_load_2d((unsigned int)(uintptr_t)&sLo[1 - cur][0],
                        gLo + (size_t)cb * DD, DD, CHUNK, DD, MM, DD);
            tdm_load_2d((unsigned int)(uintptr_t)&sT[1 - cur][0],
                        gT + cb, CHUNK, DD, MM, DD, MM);
        }

        chunk_compute(&sHi[cur][0], &sLo[cur][0], &sT[cur][0],
                      xh, xl, l16, hi, run_m, run_s, o);
    }

    store_out(Out, q, hi, run_s, o);
}

// ---------------------------------------------------------------------------
// Fallback (no workspace): in-kernel conversion, single-buffered
// ---------------------------------------------------------------------------
__global__ __launch_bounds__(256, 1)
void attn_read_fallback(const float* __restrict__ X,
                        const float* __restrict__ Mem,
                        float* __restrict__ Out) {
    __shared__ __align__(16) _Float16 sHi[CHUNK * DD];
    __shared__ __align__(16) _Float16 sLo[CHUNK * DD];
    __shared__ __align__(16) _Float16 sT [DD * CHUNK];

    const int tid  = threadIdx.x;
    const int lane = tid & 31;
    const int wave = tid >> 5;
    const int l16  = lane & 15;
    const int hi   = lane >> 4;
    const int q    = blockIdx.x * 128 + wave * 16 + l16;

    v16h xh[4], xl[4];
    load_x_frags(X + (size_t)q * DD, hi, xh, xl);

    v8f o[8];
    #pragma unroll
    for (int t = 0; t < 8; ++t)
        #pragma unroll
        for (int e = 0; e < 8; ++e) o[t][e] = 0.0f;

    float run_m = -INFINITY, run_s = 0.0f;

    for (int cb = 0; cb < MM; cb += CHUNK) {
        #pragma unroll
        for (int k = 0; k < 4; ++k) {
            int idx = tid + k * 256;
            int m = idx >> 5, c4 = idx & 31;
            v4f f = *(const v4f*)(Mem + (size_t)(cb + m) * DD + c4 * 4);
            v4h vh, vl;
            #pragma unroll
            for (int c = 0; c < 4; ++c) {
                _Float16 h2 = (_Float16)f[c];
                vh[c] = h2;
                vl[c] = (_Float16)(f[c] - (float)h2);
                sT[(c4 * 4 + c) * CHUNK + m] = h2;
            }
            *(v4h*)(sHi + m * DD + c4 * 4) = vh;
            *(v4h*)(sLo + m * DD + c4 * 4) = vl;
        }
        __syncthreads();

        chunk_compute(sHi, sLo, sT, xh, xl, l16, hi, run_m, run_s, o);
        __syncthreads();
    }

    store_out(Out, q, hi, run_s, o);
}

extern "C" void kernel_launch(void* const* d_in, const int* in_sizes, int n_in,
                              void* d_out, int out_size, void* d_ws, size_t ws_size,
                              hipStream_t stream) {
    (void)in_sizes; (void)n_in; (void)out_size;
    const float* X   = (const float*)d_in[0];
    const float* Mem = (const float*)d_in[1];
    float* Out       = (float*)d_out;

    const size_t plane = (size_t)MM * DD * sizeof(_Float16);   // 1 MB
    if (ws_size >= 3 * plane) {
        _Float16* wsHi = (_Float16*)((char*)d_ws);
        _Float16* wsLo = (_Float16*)((char*)d_ws + plane);
        _Float16* wsT  = (_Float16*)((char*)d_ws + 2 * plane);
        prep_kernel<<<dim3(MM * DD / 4 / 256), dim3(256), 0, stream>>>(Mem, wsHi, wsLo, wsT);
        attn_read_tdm<<<dim3(BQ / 128), dim3(256), 0, stream>>>(X, wsHi, wsLo, wsT, Out);
    } else {
        attn_read_fallback<<<dim3(BQ / 128), dim3(256), 0, stream>>>(X, Mem, Out);
    }
}
